// FLAGatedLinearAttention_7756710936885
// MI455X (gfx1250) — compile-verified, matrix operand order checked by reading, not run on hardware
//
#include <hip/hip_runtime.h>
#include <hip/hip_bf16.h>
#include <math.h>

// ---------------- problem constants (from reference) ----------------
#define B_ 2
#define T_ 2048
#define D_ 1024
#define H_ 4
#define KD_ 512
#define VD_ 1024
#define DK_ 128
#define DV_ 256
#define CHUNK_ 64
#define NC_ 32
#define LOWR_ 16
#define DVB_ 128   // dv columns handled per workgroup in the scan
#define NVB_ 2     // DV_/DVB_

// ---------------- WMMA plumbing (CDNA5, wave32) ----------------
typedef __attribute__((ext_vector_type(16))) __bf16 v16bf;
typedef __attribute__((ext_vector_type(2)))  __bf16 v2bf;
typedef __attribute__((ext_vector_type(8)))  float  v8f;

// fast f32 -> bf16 (truncate): single v_lshrrev
__device__ __forceinline__ unsigned short f2bf(float f) {
  return (unsigned short)(__float_as_uint(f) >> 16);
}

// pack two f32 -> two bf16 in one dword:
//  1) hardware packed convert if the builtin exists (RNE, 1 op)
//  2) v_perm_b32 byte-select of the two high halves (truncate, 1 op)
#if __has_builtin(__builtin_amdgcn_cvt_pk_bf16_f32)
__device__ __forceinline__ unsigned int f2bf2(float lo, float hi) {
  union { v2bf v; unsigned int u; } cv;
  cv.v = __builtin_amdgcn_cvt_pk_bf16_f32(lo, hi);
  return cv.u;
}
#elif __has_builtin(__builtin_amdgcn_perm)
__device__ __forceinline__ unsigned int f2bf2(float lo, float hi) {
  // result bytes: {hi.b3, hi.b2, lo.b3, lo.b2} -> {hi[31:16], lo[31:16]}
  return __builtin_amdgcn_perm(__float_as_uint(hi), __float_as_uint(lo), 0x07060302u);
}
#else
__device__ __forceinline__ unsigned int f2bf2(float lo, float hi) {
  return (unsigned int)f2bf(lo) | ((unsigned int)f2bf(hi) << 16);
}
#endif

union F16x {
  v16bf v;
  uint4 q[2];
  unsigned int d[8];
  unsigned short u[16];
};

__device__ __forceinline__ v16bf ldFrag(const unsigned short* p) {
  // 32B aligned LDS read: 16 contiguous bf16 = one lane's share of a 16x32 operand
  F16x f;
  const uint4* q = (const uint4*)p;
  f.q[0] = q[0];
  f.q[1] = q[1];
  return f.v;
}

__device__ __forceinline__ v8f wmma_bf16(v16bf a, v16bf b, v8f c) {
  return __builtin_amdgcn_wmma_f32_16x16x32_bf16(false, a, false, b, (short)0, c,
                                                 false, false);
}

// ---------------- generic GEMM: C[M,N] = A[M,K] @ W[K,N], f32 in/out ----------------
// 128x128 block tile, K-step 32, double-buffered LDS, 8 waves in a 4x2 grid,
// each wave owns a 32x64 subtile (2 M-tiles x 4 N-tiles -> 8 WMMAs / k-step).
__global__ __launch_bounds__(256) void gemm_xw_bf16(const float* __restrict__ A,
                                                    const float* __restrict__ W,
                                                    float* __restrict__ C,
                                                    int M, int N, int K) {
  __shared__ unsigned int aP[2][128][16];  // [buf][m][k-pair], row-major in k
  __shared__ unsigned int bP[2][128][16];  // [buf][n][k-pair], N-major (transposed)
  const int tid = threadIdx.x;
  const int w = tid >> 5, lane = tid & 31, lm = lane & 15, lh = lane >> 4;
  const int wm = (w >> 1) * 2;    // first of 2 M-tiles
  const int wn = (w & 1) * 4;     // first of 4 N-tiles
  const int bn = blockIdx.x * 128, bm = blockIdx.y * 128;

  v8f acc[2][4];
#pragma unroll
  for (int i = 0; i < 2; ++i)
#pragma unroll
    for (int j = 0; j < 4; ++j) acc[i][j] = {};

  const int arow = tid >> 1, akp = (tid & 1) * 8;   // A tile: 128 rows x 32 k
  const int bk2 = (tid >> 4) * 2, bnh = (tid & 15) * 8;  // W tile: 32 k x 128 n

  float4 av[4], wv[4];
  auto loadNext = [&](int kt) {
    const float4* ap = (const float4*)(A + (size_t)(bm + arow) * K + kt + akp * 2);
#pragma unroll
    for (int j = 0; j < 4; ++j) av[j] = ap[j];
    const float* w0 = W + (size_t)(kt + bk2) * N + bn + bnh;
    const float* w1 = w0 + N;
    wv[0] = ((const float4*)w0)[0];
    wv[1] = ((const float4*)w0)[1];
    wv[2] = ((const float4*)w1)[0];
    wv[3] = ((const float4*)w1)[1];
  };
  auto stage = [&](int pp) {
#pragma unroll
    for (int j = 0; j < 4; ++j) {
      aP[pp][arow][akp + 2 * j]     = f2bf2(av[j].x, av[j].y);
      aP[pp][arow][akp + 2 * j + 1] = f2bf2(av[j].z, av[j].w);
    }
    const float* r0 = (const float*)&wv[0];  // 8 n-values at k = bk2
    const float* r1 = (const float*)&wv[2];  // 8 n-values at k = bk2+1
#pragma unroll
    for (int j = 0; j < 8; ++j)
      bP[pp][bnh + j][bk2 >> 1] = f2bf2(r0[j], r1[j]);
  };

  loadNext(0);
  stage(0);
  int p = 0;
  for (int kt = 0; kt < K; kt += 32) {
    const bool more = (kt + 32) < K;
    if (more) loadNext(kt + 32);            // global loads overlap this tile's WMMAs
    __syncthreads();
    v16bf af0 = ldFrag((const unsigned short*)&aP[p][(wm + 0) * 16 + lm][lh * 8]);
    v16bf af1 = ldFrag((const unsigned short*)&aP[p][(wm + 1) * 16 + lm][lh * 8]);
#pragma unroll
    for (int nj = 0; nj < 4; ++nj) {
      v16bf bf = ldFrag((const unsigned short*)&bP[p][(wn + nj) * 16 + lm][lh * 8]);
      acc[0][nj] = wmma_bf16(af0, bf, acc[0][nj]);
      acc[1][nj] = wmma_bf16(af1, bf, acc[1][nj]);
    }
    if (more) stage(p ^ 1);                 // stores go to the other buffer
    p ^= 1;
  }

#pragma unroll
  for (int i = 0; i < 2; ++i)
#pragma unroll
    for (int nj = 0; nj < 4; ++nj)
#pragma unroll
      for (int r = 0; r < 8; ++r) {
        int row = bm + (wm + i) * 16 + r + lh * 8;
        int col = bn + (wn + nj) * 16 + lm;
        C[(size_t)row * N + col] = acc[i][nj][r];
      }
}

// ---------------- low-rank projection: low[M,16] = x @ Wgk1 (K too skinny for WMMA) ----
__global__ __launch_bounds__(128) void lowrank_proj(const float* __restrict__ x,
                                                    const float* __restrict__ Wgk1,
                                                    float* __restrict__ low) {
  __shared__ float red[128];
  const int row = blockIdx.x;
  const int tid = threadIdx.x;
  const int r = tid & 15, seg = tid >> 4;  // 8 segments of 128 along D
  const float* xp = x + (size_t)row * D_ + seg * 128;
  float s = 0.f;
  for (int d = 0; d < 128; ++d) s += xp[d] * Wgk1[(size_t)(seg * 128 + d) * LOWR_ + r];
  red[tid] = s;
  __syncthreads();
  if (tid < 16) {
    float t = 0.f;
    for (int sgi = 0; sgi < 8; ++sgi) t += red[sgi * 16 + tid];
    low[(size_t)row * LOWR_ + tid] = t;
  }
}

// ---------------- gk = log_sigmoid(low @ Wgk2 + b) / 16 ----------------
__global__ __launch_bounds__(256) void gk_proj(const float* __restrict__ low,
                                               const float* __restrict__ Wgk2,
                                               const float* __restrict__ bgk2,
                                               float* __restrict__ gk) {
  size_t i = (size_t)blockIdx.x * blockDim.x + threadIdx.x;
  if (i >= (size_t)B_ * T_ * KD_) return;
  int kd = (int)(i % KD_);
  size_t t = i / KD_;
  float z = bgk2[kd];
#pragma unroll
  for (int r = 0; r < LOWR_; ++r) z += low[t * LOWR_ + r] * Wgk2[r * KD_ + kd];
  float ls = fminf(z, 0.f) - log1pf(__expf(-fabsf(z)));   // log_sigmoid
  gk[i] = ls * (1.0f / 16.0f);                            // / NORMALIZER
}

// ---------------- chunked gated linear attention scan ----------------
// grid = B*H*NVB workgroups; each owns (b,h) and a 128-wide dv slice.
// State S[128][128] f32 lives in LDS across the 32 sequential chunks.
__global__ __launch_bounds__(256) void fla_chunk_scan(const float* __restrict__ qw,
                                                      const float* __restrict__ kw,
                                                      const float* __restrict__ vw,
                                                      const float* __restrict__ gkw,
                                                      float* __restrict__ ow) {
  extern __shared__ char smem[];
  float* S             = (float*)(smem);                   // 128*128 f32  (64KB)
  float* Gc            = (float*)(smem + 65536);           // 64*128 f32   (32KB)
  unsigned int* qgP    = (unsigned int*)(smem + 98304);    // 64*128 bf16  (A layout)
  unsigned int* kgP    = (unsigned int*)(smem + 114688);   // 64*128 bf16  (B^T layout)
  unsigned int* kbTP   = (unsigned int*)(smem + 131072);   // 128*64 bf16  (k_bar^T rows)
  unsigned int* vTP    = (unsigned int*)(smem + 147456);   // 128*64 bf16  (v^T rows)
  unsigned short* Am   = (unsigned short*)(smem + 163840); // 64*64 bf16   (masked A)
  float* Glast         = (float*)(smem + 172032);          // 128 f32
  const unsigned short* qg  = (const unsigned short*)qgP;
  const unsigned short* kg  = (const unsigned short*)kgP;
  const unsigned short* kbT = (const unsigned short*)kbTP;
  const unsigned short* vT  = (const unsigned short*)vTP;

  const int tid = threadIdx.x;
  const int w = tid >> 5, lane = tid & 31, lm = lane & 15, lh = lane >> 4;
  const int bid = blockIdx.x;
  const int b  = bid / (H_ * NVB_);
  const int h  = (bid / NVB_) % H_;
  const int vb = bid % NVB_;
  const float scale = 0.08838834764831845f;  // DK^-0.5

  for (int i = tid; i < DK_ * DVB_; i += 256) S[i] = 0.f;
  __syncthreads();

  for (int n = 0; n < NC_; ++n) {
    const int t0 = n * CHUNK_;
    const size_t rowQ = ((size_t)b * T_ + t0) * KD_ + h * DK_;
    const size_t rowV = ((size_t)b * T_ + t0) * VD_ + h * DV_ + vb * DVB_;

    // ---- phase A: cumsum G, build bf16 operand tiles (packed dword LDS stores) ----
    for (int i = tid; i < CHUNK_ * CHUNK_; i += 256) Am[i] = 0;  // zero incl. upper tri
    if (tid < DK_) {
      float s = 0.f;
      const float* gp = gkw + rowQ + tid;
      for (int c = 0; c < CHUNK_; ++c) {
        s += gp[(size_t)c * KD_];
        Gc[c * DK_ + tid] = s;
      }
      Glast[tid] = s;
    }
    __syncthreads();
    // qg/kg: pairs along dk (row-major layout)
    for (int i = tid; i < CHUNK_ * DK_ / 2; i += 256) {
      int c = i >> 6, dk2 = (i & 63) * 2;
      float2 qv = *(const float2*)(qw + rowQ + (size_t)c * KD_ + dk2);
      float2 kv = *(const float2*)(kw + rowQ + (size_t)c * KD_ + dk2);
      float G0 = Gc[c * DK_ + dk2], G1 = Gc[c * DK_ + dk2 + 1];
      qgP[c * 64 + (dk2 >> 1)] = f2bf2(qv.x * __expf(G0) * scale, qv.y * __expf(G1) * scale);
      kgP[c * 64 + (dk2 >> 1)] = f2bf2(kv.x * __expf(-G0), kv.y * __expf(-G1));
    }
    // k_bar^T: pairs along c (transposed layout)
    for (int i = tid; i < DK_ * CHUNK_ / 2; i += 256) {
      int dk = i & 127, c2 = (i >> 7) * 2;
      float k0 = kw[rowQ + (size_t)c2 * KD_ + dk];
      float k1 = kw[rowQ + (size_t)(c2 + 1) * KD_ + dk];
      float gl = Glast[dk];
      kbTP[dk * 32 + (c2 >> 1)] = f2bf2(k0 * __expf(gl - Gc[c2 * DK_ + dk]),
                                        k1 * __expf(gl - Gc[(c2 + 1) * DK_ + dk]));
    }
    // v^T: pairs along c
    for (int i = tid; i < DVB_ * CHUNK_ / 2; i += 256) {
      int dv = i & 127, c2 = (i >> 7) * 2;
      float v0 = vw[rowV + (size_t)c2 * VD_ + dv];
      float v1 = vw[rowV + (size_t)(c2 + 1) * VD_ + dv];
      vTP[dv * 32 + (c2 >> 1)] = f2bf2(v0, v1);
    }
    __syncthreads();

    // ---- phase B: A = tril(qg @ kg^T), 10 lower-tri 16x16 tiles over 8 waves ----
    {
      int cnt = 0;
#pragma unroll
      for (int mi = 0; mi < 4; ++mi)
#pragma unroll
        for (int nj = 0; nj <= mi; ++nj, ++cnt) {
          if ((cnt & 7) != w) continue;
          v8f acc = {};
#pragma unroll
          for (int ks = 0; ks < 4; ++ks) {
            int dk0 = ks * 32 + lh * 16;
            v16bf af = ldFrag(&qg[(mi * 16 + lm) * DK_ + dk0]);
            v16bf bf = ldFrag(&kg[(nj * 16 + lm) * DK_ + dk0]);
            acc = wmma_bf16(af, bf, acc);
          }
#pragma unroll
          for (int r = 0; r < 8; ++r) {
            int i2 = mi * 16 + r + lh * 8;
            int j2 = nj * 16 + lm;
            Am[i2 * CHUNK_ + j2] = (i2 >= j2) ? f2bf(acc[r]) : (unsigned short)0;
          }
        }
    }
    __syncthreads();

    // ---- phase C: o = qg @ S_prev + Am @ v; wave w owns dv cols [16w,16w+16) ----
    {
      v8f acc[4] = {{}, {}, {}, {}};
      for (int ks = 0; ks < 4; ++ks) {                      // inter-chunk, K = DK = 128
        int dk0 = ks * 32 + lh * 16;
        int col = w * 16 + lm;
        F16x sb;
#pragma unroll
        for (int j = 0; j < 8; ++j)                         // packed converts, col of S
          sb.d[j] = f2bf2(S[(dk0 + 2 * j) * DVB_ + col], S[(dk0 + 2 * j + 1) * DVB_ + col]);
#pragma unroll
        for (int mi = 0; mi < 4; ++mi) {
          v16bf af = ldFrag(&qg[(mi * 16 + lm) * DK_ + dk0]);
          acc[mi] = wmma_bf16(af, sb.v, acc[mi]);
        }
      }
#pragma unroll
      for (int ks = 0; ks < 2; ++ks) {                      // intra-chunk, K = CHUNK = 64
        int c0 = ks * 32 + lh * 16;
        v16bf bvf = ldFrag(&vT[(w * 16 + lm) * CHUNK_ + c0]);
#pragma unroll
        for (int mi = 0; mi < 4; ++mi) {
          v16bf af = ldFrag(&Am[(mi * 16 + lm) * CHUNK_ + c0]);
          acc[mi] = wmma_bf16(af, bvf, acc[mi]);
        }
      }
#pragma unroll
      for (int mi = 0; mi < 4; ++mi)
#pragma unroll
        for (int r = 0; r < 8; ++r) {
          int tt = t0 + mi * 16 + r + lh * 8;
          int dv = h * DV_ + vb * DVB_ + w * 16 + lm;
          ow[((size_t)b * T_ + tt) * VD_ + dv] = acc[mi][r];
        }
    }
    __syncthreads();   // all reads of S_prev complete before update

    // ---- phase D: S = diag(exp(Glast)) S + k_bar^T @ v; wave w owns dk rows [16w,16w+16) ----
    {
#pragma unroll
      for (int ni = 0; ni < 8; ++ni) {
        v8f acc = {};
#pragma unroll
        for (int ks = 0; ks < 2; ++ks) {
          int c0 = ks * 32 + lh * 16;
          v16bf af = ldFrag(&kbT[(w * 16 + lm) * CHUNK_ + c0]);
          v16bf bf = ldFrag(&vT[(ni * 16 + lm) * CHUNK_ + c0]);
          acc = wmma_bf16(af, bf, acc);
        }
#pragma unroll
        for (int r = 0; r < 8; ++r) {
          int m  = w * 16 + r + lh * 8;
          int cc = ni * 16 + lm;
          S[m * DVB_ + cc] = __expf(Glast[m]) * S[m * DVB_ + cc] + acc[r];
        }
      }
    }
    __syncthreads();
  }
}

// ---------------- group RMSNorm + SiLU gate ----------------
__global__ __launch_bounds__(64) void norm_gate(const float* __restrict__ ow,
                                                const float* __restrict__ gw,
                                                const float* __restrict__ gnw,
                                                float* __restrict__ y) {
  __shared__ float red[64];
  const int bid = blockIdx.x;          // b*T*H
  const int h = bid % H_;
  const size_t bt = bid / H_;
  const int tid = threadIdx.x;
  const float* op = ow + bt * VD_ + h * DV_;
  const float* gp = gw + bt * VD_ + h * DV_;
  float ov[4];
  float ss = 0.f;
#pragma unroll
  for (int j = 0; j < 4; ++j) {
    ov[j] = op[tid + j * 64];
    ss += ov[j] * ov[j];
  }
  red[tid] = ss;
  __syncthreads();
  for (int s = 32; s > 0; s >>= 1) {
    if (tid < s) red[tid] += red[tid + s];
    __syncthreads();
  }
  float inv = rsqrtf(red[0] * (1.0f / (float)DV_) + 1e-6f);
  float* yp = y + bt * VD_ + h * DV_;
#pragma unroll
  for (int j = 0; j < 4; ++j) {
    int dv = tid + j * 64;
    float g = gp[dv];
    float sig = 1.f / (1.f + __expf(-g));
    yp[dv] = ov[j] * inv * gnw[dv] * (g * sig);
  }
}

// ---------------- host-side launch ----------------
extern "C" void kernel_launch(void* const* d_in, const int* in_sizes, int n_in,
                              void* d_out, int out_size, void* d_ws, size_t ws_size,
                              hipStream_t stream) {
  (void)in_sizes; (void)n_in; (void)out_size; (void)ws_size;
  const float* x    = (const float*)d_in[0];
  const float* Wq   = (const float*)d_in[1];
  const float* Wk   = (const float*)d_in[2];
  const float* Wv   = (const float*)d_in[3];
  const float* Wg   = (const float*)d_in[4];
  const float* Wgk1 = (const float*)d_in[5];
  const float* Wgk2 = (const float*)d_in[6];
  const float* bgk2 = (const float*)d_in[7];
  const float* gnw  = (const float*)d_in[8];
  const float* Wo   = (const float*)d_in[9];
  float* out = (float*)d_out;

  char* ws = (char*)d_ws;
  float* q_ws   = (float*)(ws + 0);          // 4096*512*4  = 8 MB
  float* k_ws   = (float*)(ws + 8388608);    // 8 MB
  float* v_ws   = (float*)(ws + 16777216);   // 16 MB
  float* g_ws   = (float*)(ws + 33554432);   // 16 MB
  float* gk_ws  = (float*)(ws + 50331648);   // 8 MB
  float* low_ws = (float*)(ws + 58720256);   // 256 KB
  float* o_ws   = (float*)(ws + 58982400);   // 16 MB
  float* y_ws   = (float*)(ws + 75759616);   // 16 MB

  const int M = B_ * T_;  // 4096
  dim3 blk(256);
  gemm_xw_bf16<<<dim3(KD_ / 128, M / 128), blk, 0, stream>>>(x, Wq, q_ws, M, KD_, D_);
  gemm_xw_bf16<<<dim3(KD_ / 128, M / 128), blk, 0, stream>>>(x, Wk, k_ws, M, KD_, D_);
  gemm_xw_bf16<<<dim3(VD_ / 128, M / 128), blk, 0, stream>>>(x, Wv, v_ws, M, VD_, D_);
  gemm_xw_bf16<<<dim3(VD_ / 128, M / 128), blk, 0, stream>>>(x, Wg, g_ws, M, VD_, D_);
  lowrank_proj<<<M, 128, 0, stream>>>(x, Wgk1, low_ws);
  gk_proj<<<(M * KD_ + 255) / 256, 256, 0, stream>>>(low_ws, Wgk2, bgk2, gk_ws);
  fla_chunk_scan<<<B_ * H_ * NVB_, 256, 172544, stream>>>(q_ws, k_ws, v_ws, gk_ws, o_ws);
  norm_gate<<<M * H_, 64, 0, stream>>>(o_ws, g_ws, gnw, y_ws);
  gemm_xw_bf16<<<dim3(D_ / 128, M / 128), blk, 0, stream>>>(y_ws, Wo, out, M, D_, D_);
}
